// RNNModel_51110110822603
// MI455X (gfx1250) — compile-verified, matrix operand order checked by reading, not run on hardware
//
#include <hip/hip_runtime.h>
#include <hip/hip_bf16.h>
#include <math.h>

typedef __attribute__((ext_vector_type(16))) _Float16 v16h;
typedef __attribute__((ext_vector_type(8)))  float    v8f;
typedef __attribute__((vector_size(4 * sizeof(int)))) int v4i_;

#define B_  128
#define T_  256
#define V_  256
#define U_  2048

// ---------------- workspace layout (bytes) ----------------
#define OFF_BAR  0
#define OFF_WHT  256
#define WHT_BYTES ((size_t)U_ * U_ * 2)
#define OFF_WDT  (OFF_WHT + WHT_BYTES)
#define WDT_BYTES ((size_t)U_ * V_ * 2)
#define OFF_H    (OFF_WDT + WDT_BYTES)
#define H_ELEMS  ((size_t)B_ * U_)
#define OFF_XP   (OFF_H + 2 * H_ELEMS * 2)
#define XP_BYTES ((size_t)B_ * T_ * U_ * 4)
#define OFF_HS   (OFF_XP + XP_BYTES)

// CDNA5 async LDS DMA (GLOBAL_LOAD_ASYNC_TO_LDS_B128, ASYNCcnt-tracked).
#if defined(__AMDGCN__) && __has_builtin(__builtin_amdgcn_global_load_async_to_lds_b128)
#define ASYNC_LDS 1
#else
#define ASYNC_LDS 0
#endif

// Copy 16 bytes global -> LDS (async DMA when available, else via VGPRs).
__device__ __forceinline__ void cp16(_Float16* l, const _Float16* g) {
#if ASYNC_LDS
    __builtin_amdgcn_global_load_async_to_lds_b128((v4i_*)g, (v4i_*)l, 0, 0);
#else
    *(uint4*)l = *(const uint4*)g;
#endif
}

__device__ __forceinline__ void async_wait0() {
#if ASYNC_LDS
#if __has_builtin(__builtin_amdgcn_s_wait_asynccnt)
    __builtin_amdgcn_s_wait_asynccnt(0);
#else
    asm volatile("s_wait_asynccnt 0x0" ::: "memory");
#endif
#endif
}

// Load one 16x32 (or 32x16) f16 WMMA fragment row for this lane.
// Layout (ISA 7.12.2, 16-bit A/B): VGPR j<4 holds K = half*8 + 2j,2j+1;
// j>=4 holds K = 16 + half*8 + ... -> two contiguous 16B chunks.
__device__ __forceinline__ v16h load_frag(const _Float16* __restrict__ base, int half) {
    union { v16h v; uint4 q[2]; } f;
    f.q[0] = *(const uint4*)(base + half * 8);
    f.q[1] = *(const uint4*)(base + 16 + half * 8);
    return f.v;
}

__device__ __forceinline__ v8f wmma_f16(v16h a, v16h b, v8f c) {
    return __builtin_amdgcn_wmma_f32_16x16x32_f16(false, a, false, b, (short)0, c,
                                                  false, false);
}

// ---------------- kernels ----------------

// x_proj[(b*T+t)*U+u] = Wx[x[b,t]][u] + bias[u]
__global__ void k_embed(const int* __restrict__ x, const float* __restrict__ Wx,
                        const float* __restrict__ bias, float* __restrict__ xp) {
    size_t idx = (size_t)blockIdx.x * 256 + threadIdx.x;   // < B*T*U
    int u  = (int)(idx % U_);
    int bt = (int)(idx / U_);
    int tok = x[bt];
    xp[idx] = Wx[(size_t)tok * U_ + u] + bias[u];
}

// WhT[c][r] = f16(Wh[r][c])
__global__ void k_convWh(const float* __restrict__ Wh, _Float16* __restrict__ WhT) {
    size_t i = (size_t)blockIdx.x * 256 + threadIdx.x;     // < U*U
    int r = (int)(i / U_);
    int c = (int)(i % U_);
    WhT[(size_t)c * U_ + r] = (_Float16)Wh[i];
}

// WdT[v][u] = f16(Wd[u][v])
__global__ void k_convWd(const float* __restrict__ Wd, _Float16* __restrict__ WdT) {
    size_t i = (size_t)blockIdx.x * 256 + threadIdx.x;     // < U*V
    int u = (int)(i / V_);
    int v = (int)(i % V_);
    WdT[(size_t)v * U_ + u] = (_Float16)Wd[i];
}

// Hbuf[0] = f16(h0); zero grid-barrier counter
__global__ void k_init(const float* __restrict__ h0, _Float16* __restrict__ Hbuf,
                       unsigned* __restrict__ bar) {
    size_t i = (size_t)blockIdx.x * 256 + threadIdx.x;     // < B*U
    Hbuf[i] = (_Float16)h0[i];
    if (i == 0) *bar = 0u;
}

// Persistent recurrence: 64 blocks x 128 threads (4 waves), block tile 64x64,
// wave tile 32x32 (2x2 WMMA tiles). A/B k-slabs are staged into LDS with
// double buffering: async loads for slab k+1 are issued before the WMMAs of
// slab k, so the DMA overlaps the matrix math; one barrier per k-iteration.
// Per step: H_new = tanh(Xproj_t + H_prev @ Wh), f16 WMMA, f32 accumulate.
__global__ void __launch_bounds__(128) k_rnn(
        const float* __restrict__ xp, const _Float16* __restrict__ WhT,
        _Float16* __restrict__ Hbuf, _Float16* __restrict__ hs,
        unsigned* __restrict__ bar) {
    __shared__ _Float16 As[2][64 * 32];   // [buf][row][k]  rows = batch rows
    __shared__ _Float16 Bs[2][64 * 32];   // [buf][col][k]  cols = output units

    const int tid  = threadIdx.x;
    const int lane = tid & 31;
    const int wid  = tid >> 5;
    const int bi   = blockIdx.x >> 5;        // 0..1   (M groups of 64)
    const int bj   = blockIdx.x & 31;        // 0..31  (N groups of 64)
    const int i0   = bi * 64;
    const int j0   = bj * 64;
    const int wm   = wid >> 1;               // wave row inside block tile
    const int wn   = wid & 1;                // wave col inside block tile
    const int m    = lane & 15;
    const int half = lane >> 4;

    // cooperative-load coordinates: 2 x 16B chunks per thread per matrix
    const int r0 = tid >> 2;                 // 0..31
    const int o0 = (tid & 3) * 8;            // f16 offset within 32-wide k slab
    const int r1 = r0 + 32;                  // 32..63

    for (int t = 0; t < T_; ++t) {
        const _Float16* Hc = Hbuf + (size_t)(t & 1) * H_ELEMS;
        _Float16*       Hn = Hbuf + (size_t)((t + 1) & 1) * H_ELEMS;

        // prologue: fill slab 0
        cp16(&As[0][r0 * 32 + o0], Hc  + (size_t)(i0 + r0) * U_ + o0);
        cp16(&As[0][r1 * 32 + o0], Hc  + (size_t)(i0 + r1) * U_ + o0);
        cp16(&Bs[0][r0 * 32 + o0], WhT + (size_t)(j0 + r0) * U_ + o0);
        cp16(&Bs[0][r1 * 32 + o0], WhT + (size_t)(j0 + r1) * U_ + o0);

        v8f acc[2][2] = {};
        for (int ki = 0; ki < U_ / 32; ++ki) {
            const int cur = ki & 1;
            async_wait0();       // this wave's DMA into slab `cur` has landed
            __syncthreads();     // everyone's landed; prev reads of `nxt` done

            if (ki + 1 < U_ / 32) {        // kick off slab k+1 DMA
                const int nxt = cur ^ 1;
                const int k0  = (ki + 1) * 32;
                cp16(&As[nxt][r0 * 32 + o0], Hc  + (size_t)(i0 + r0) * U_ + k0 + o0);
                cp16(&As[nxt][r1 * 32 + o0], Hc  + (size_t)(i0 + r1) * U_ + k0 + o0);
                cp16(&Bs[nxt][r0 * 32 + o0], WhT + (size_t)(j0 + r0) * U_ + k0 + o0);
                cp16(&Bs[nxt][r1 * 32 + o0], WhT + (size_t)(j0 + r1) * U_ + k0 + o0);
            }

            const _Float16* a_ = As[cur];
            const _Float16* b_ = Bs[cur];
            v16h a0 = load_frag(a_ + (wm * 32 +      m) * 32, half);
            v16h a1 = load_frag(a_ + (wm * 32 + 16 + m) * 32, half);
            v16h b0 = load_frag(b_ + (wn * 32 +      m) * 32, half);
            v16h b1 = load_frag(b_ + (wn * 32 + 16 + m) * 32, half);
            acc[0][0] = wmma_f16(a0, b0, acc[0][0]);
            acc[0][1] = wmma_f16(a0, b1, acc[0][1]);
            acc[1][0] = wmma_f16(a1, b0, acc[1][0]);
            acc[1][1] = wmma_f16(a1, b1, acc[1][1]);
        }

        // D layout: lane half h, VGPR r -> M = r + 8*h, N = lane&15
#pragma unroll
        for (int mt = 0; mt < 2; ++mt) {
#pragma unroll
            for (int nt = 0; nt < 2; ++nt) {
#pragma unroll
                for (int r = 0; r < 8; ++r) {
                    int bb = i0 + wm * 32 + mt * 16 + r + half * 8;
                    int uu = j0 + wn * 32 + nt * 16 + m;
                    size_t xi = ((size_t)bb * T_ + t) * U_ + uu;
                    float v = tanhf(acc[mt][nt][r] + xp[xi]);
                    _Float16 hv = (_Float16)v;
                    Hn[(size_t)bb * U_ + uu] = hv;
                    hs[xi] = hv;                 // hs row index = b*T + t
                }
            }
        }

        // grid-wide barrier: monotonic counter, zeroed each launch by k_init
        __threadfence();
        __syncthreads();
        if (tid == 0) {
            __hip_atomic_fetch_add(bar, 1u, __ATOMIC_ACQ_REL, __HIP_MEMORY_SCOPE_AGENT);
            unsigned target = (unsigned)(t + 1) * 64u;
            while (__hip_atomic_load(bar, __ATOMIC_ACQUIRE, __HIP_MEMORY_SCOPE_AGENT) < target)
                __builtin_amdgcn_s_sleep(2);
        }
        __syncthreads();
    }
}

// logits[m][v] = hs[m][:] . WdT[v][:] + bd[v],  m = b*T + t   (32768 x 256)
// Wave tile 32x64 (2M x 4N wmma tiles): A reused 4x, B (L2-resident) 2x.
__global__ void __launch_bounds__(128) k_logits(
        const _Float16* __restrict__ hs, const _Float16* __restrict__ WdT,
        const float* __restrict__ bd, float* __restrict__ out) {
    const int lane = threadIdx.x & 31;
    const int wid  = threadIdx.x >> 5;
    const int w    = blockIdx.x * 4 + wid;   // 0..4095
    const int m0   = (w >> 2) * 32;          // 1024 row groups of 32
    const int n0   = (w & 3) * 64;           // 4 col groups of 64
    const int m    = lane & 15;
    const int half = lane >> 4;

    const _Float16* a0row = hs + (size_t)(m0 + m) * U_;
    const _Float16* a1row = hs + (size_t)(m0 + 16 + m) * U_;

    v8f acc[2][4] = {};
    for (int k0 = 0; k0 < U_; k0 += 32) {
        __builtin_prefetch(a0row + k0 + 512, 0, 1);
        __builtin_prefetch(a1row + k0 + 512, 0, 1);
        v16h a0 = load_frag(a0row + k0, half);
        v16h a1 = load_frag(a1row + k0, half);
#pragma unroll
        for (int nt = 0; nt < 4; ++nt) {
            v16h b = load_frag(WdT + (size_t)(n0 + nt * 16 + m) * U_ + k0, half);
            acc[0][nt] = wmma_f16(a0, b, acc[0][nt]);
            acc[1][nt] = wmma_f16(a1, b, acc[1][nt]);
        }
    }

#pragma unroll
    for (int nt = 0; nt < 4; ++nt) {
        float bias = bd[n0 + nt * 16 + m];
#pragma unroll
        for (int mt = 0; mt < 2; ++mt) {
#pragma unroll
            for (int r = 0; r < 8; ++r) {
                int row = m0 + mt * 16 + r + half * 8;
                out[(size_t)row * V_ + n0 + nt * 16 + m] = acc[mt][nt][r] + bias;
            }
        }
    }
}

// ---------------- launch ----------------
extern "C" void kernel_launch(void* const* d_in, const int* in_sizes, int n_in,
                              void* d_out, int out_size, void* d_ws, size_t ws_size,
                              hipStream_t stream) {
    (void)in_sizes; (void)n_in; (void)out_size; (void)ws_size;
    const int*   x  = (const int*)d_in[0];
    const float* Wx = (const float*)d_in[1];
    const float* Wh = (const float*)d_in[2];
    const float* bv = (const float*)d_in[3];
    const float* Wd = (const float*)d_in[4];
    const float* bd = (const float*)d_in[5];
    const float* h0 = (const float*)d_in[6];
    float* out = (float*)d_out;

    char* ws = (char*)d_ws;
    unsigned* bar  = (unsigned*)(ws + OFF_BAR);
    _Float16* WhT  = (_Float16*)(ws + OFF_WHT);
    _Float16* WdT  = (_Float16*)(ws + OFF_WDT);
    _Float16* Hbuf = (_Float16*)(ws + OFF_H);
    float*    xp   = (float*)(ws + OFF_XP);
    _Float16* hs   = (_Float16*)(ws + OFF_HS);

    k_init  <<<(B_ * U_) / 256, 256, 0, stream>>>(h0, Hbuf, bar);
    k_convWh<<<(U_ * U_) / 256, 256, 0, stream>>>(Wh, WhT);
    k_convWd<<<(U_ * V_) / 256, 256, 0, stream>>>(Wd, WdT);
    k_embed <<<(B_ * T_ * U_) / 256, 256, 0, stream>>>(x, Wx, bv, xp);
    k_rnn   <<<64, 128, 0, stream>>>(xp, WhT, Hbuf, hs, bar);
    k_logits<<<1024, 128, 0, stream>>>(hs, WdT, bd, out);
}
